// EfficientAttention_43989055045946
// MI455X (gfx1250) — matrix operand order repaired and checked by reading, not compiled, not run on hardware
//
#include <hip/hip_runtime.h>
#include <hip/hip_bf16.h>

// ---------------------------------------------------------------------------
// Efficient attention (Shen et al. style), fully fused pipeline for gfx1250.
// GEMMs: v_wmma_f32_16x16x32_bf16 (wave32 WMMA, f32 accumulate).
// Data movement: global_load_async_to_lds_b128 (ASYNCcnt) for bf16 staging,
// tensor_load_to_lds (TDM, TENSORcnt) for the proj A-tile.
// ---------------------------------------------------------------------------

typedef __attribute__((ext_vector_type(16))) __bf16 v16bf;
typedef __attribute__((ext_vector_type(8)))  __bf16 v8bf;
typedef __attribute__((ext_vector_type(8)))  float  v8f;
typedef __attribute__((ext_vector_type(4)))  unsigned int u32x4;
typedef __attribute__((ext_vector_type(8)))  int  i32x8;
typedef __attribute__((ext_vector_type(4)))  int  i32x4;

#define DIMD   384
#define QKV3   1152
#define NTOK   16384
#define BATCH  8
#define HEADS  8
#define CH     48
#define MTOT   (BATCH * NTOK)      // 131072 token rows
#define NCHUNK 8
#define CHUNK  (NTOK / NCHUNK)     // 2048 tokens per split-K chunk

// Workspace layout (bytes). Total ~709 MB.
static constexpr size_t OFF_Q    = 0;                        // q pre-softmax  f32 [B,H,N,48]
static constexpr size_t OFF_KT   = OFF_Q    + 201326592ull;  // k transposed   f32 [B,H,48,N]
static constexpr size_t OFF_V    = OFF_KT   + 201326592ull;  // v              bf16[B,H,N,48]
static constexpr size_t OFF_QPAD = OFF_V    + 100663296ull;  // q softmaxed    bf16[B,H,N,64]
static constexpr size_t OFF_STAT = OFF_QPAD + 134217728ull;  // k stats        f32x2 [B*H*48]
static constexpr size_t OFF_CTXP = OFF_STAT + 32768ull;      // ctx partials   f32 [B*H,8,48,48]
static constexpr size_t OFF_CTXB = OFF_CTXP + 4718592ull;    // ctx            bf16[B*H,64,48]
static constexpr size_t OFF_ATTN = OFF_CTXB + 393216ull;     // attn out       bf16[B,N,384]

// --- CDNA5 helpers ---------------------------------------------------------
__device__ __forceinline__ void async_b128_to_lds(uint32_t lds_off, const void* gaddr) {
    // per-lane global -> LDS copy, tracked with ASYNCcnt
    asm volatile("global_load_async_to_lds_b128 %0, %1, off"
                 :: "v"(lds_off), "v"(gaddr) : "memory");
}
__device__ __forceinline__ void wait_asynccnt0() {
    asm volatile("s_wait_asynccnt 0x0" ::: "memory");
}

#if __has_builtin(__builtin_amdgcn_tensor_load_to_lds)
#define HAVE_TDM 1
// 2D bf16 tile load via the Tensor Data Mover. One issue per wave.
// This toolchain exposes the 6-arg builtin:
//   (u32x4 g0, i32x8 g1, i32x4 g2, i32x4 g3, i32x8 g4, i32 cpol)
__device__ __forceinline__ void tdm_load_2d_bf16(
    uint32_t lds_byte_off, const void* gaddr,
    uint32_t tile_w, uint32_t tile_h,            // in elements
    uint32_t row_stride,                          // elements between rows
    uint32_t pad_interval_code, uint32_t pad_amount_code, bool pad_en)
{
    uint64_t ga = (uint64_t)(uintptr_t)gaddr;
    u32x4 g0;
    g0.x = 1u;                                    // count=1, user descriptor
    g0.y = lds_byte_off;                          // lds_addr
    g0.z = (uint32_t)ga;                          // global_addr[31:0]
    g0.w = (uint32_t)((ga >> 32) & 0x01FFFFFFull) | (2u << 30);   // addr[56:32], type=2
    i32x8 g1;
    g1[0] = (int)((1u << 16) |                    // data_size = 1 (2-byte elems)
                  (pad_en ? ((1u << 20) | (pad_interval_code << 22)
                                        | (pad_amount_code   << 25)) : 0u));
    g1[1] = (int)((tile_w & 0xFFFFu) << 16);      // tensor_dim0[15:0]
    g1[2] = (int)(((tile_w >> 16) & 0xFFFFu) | ((tile_h & 0xFFFFu) << 16));   // dim0 hi | dim1 lo
    g1[3] = (int)(((tile_h >> 16) & 0xFFFFu) | ((tile_w & 0xFFFFu) << 16));   // dim1 hi | tile_dim0
    g1[4] = (int)(tile_h & 0xFFFFu);              // tile_dim1 (tile_dim2 = 0)
    g1[5] = (int)row_stride;                      // tensor_dim0_stride[31:0]
    g1[6] = 0;
    g1[7] = 0;
    i32x4 zz4 = {};
    i32x8 zz8 = {};
    __builtin_amdgcn_tensor_load_to_lds(g0, g1, zz4, zz4, zz8, 0);
}
#else
#define HAVE_TDM 0
#endif

// ===========================================================================
// K1: qkv = x @ W_qkv   (M=131072, N=1152, K=384), bf16 WMMA, routed epilogue
// block 256 = 8 waves (4x2), tile 128x64, BK=32
// ===========================================================================
__global__ __launch_bounds__(256) void k_qkv_gemm(
    const float* __restrict__ x, const float* __restrict__ Wqkv,
    float* __restrict__ q_buf, float* __restrict__ kT_buf,
    __bf16* __restrict__ v_buf)
{
    __shared__ __bf16 ldsA[128][40];   // [m][k], padded stride keeps 16B align
    __shared__ __bf16 ldsB[64][40];    // [n][k] (transposed on store)

    const int t    = threadIdx.x;
    const int lane = t & 31;
    const int wave = t >> 5;
    const int wm   = wave & 3;
    const int wn   = wave >> 2;
    const int tileM = blockIdx.x * 128;
    const int tileN = blockIdx.y * 64;

    const v8f vzero = {};
    v8f acc[2][2];
    for (int i = 0; i < 2; ++i) for (int j = 0; j < 2; ++j) acc[i][j] = vzero;

    for (int k0 = 0; k0 < DIMD; k0 += 32) {
        // ---- load phase: all global loads issued with independent regs ----
        float4 fa[4], fb[2];
        #pragma unroll
        for (int q = 0; q < 4; ++q) {
            int fid  = t + 256 * q;            // 0..1023
            int row  = fid >> 3;
            int colv = fid & 7;
            fa[q] = *(const float4*)(x + (size_t)(tileM + row) * DIMD + k0 + colv * 4);
        }
        #pragma unroll
        for (int q = 0; q < 2; ++q) {
            int fid  = t + 256 * q;            // 0..511
            int row  = fid >> 4;
            int colv = fid & 15;
            fb[q] = *(const float4*)(Wqkv + (size_t)(k0 + row) * QKV3 + tileN + colv * 4);
        }
        if (k0 + 32 < DIMD) {                  // prefetch next K tile
            __builtin_prefetch(x + (size_t)(tileM + (t >> 3)) * DIMD + k0 + 32 + (t & 7) * 4, 0, 1);
            __builtin_prefetch(Wqkv + (size_t)(k0 + 32 + (t >> 4)) * QKV3 + tileN + (t & 15) * 4, 0, 1);
        }
        // ---- convert + LDS store phase ----
        #pragma unroll
        for (int q = 0; q < 4; ++q) {
            int fid  = t + 256 * q;
            int row  = fid >> 3;
            int colv = fid & 7;
            __bf16* dst = &ldsA[row][colv * 4];
            dst[0] = (__bf16)fa[q].x; dst[1] = (__bf16)fa[q].y;
            dst[2] = (__bf16)fa[q].z; dst[3] = (__bf16)fa[q].w;
        }
        #pragma unroll
        for (int q = 0; q < 2; ++q) {
            int fid  = t + 256 * q;
            int row  = fid >> 4;
            int colv = fid & 15;
            ldsB[colv * 4 + 0][row] = (__bf16)fb[q].x;
            ldsB[colv * 4 + 1][row] = (__bf16)fb[q].y;
            ldsB[colv * 4 + 2][row] = (__bf16)fb[q].z;
            ldsB[colv * 4 + 3][row] = (__bf16)fb[q].w;
        }
        __syncthreads();

        const int kbA = (lane >> 4) * 8;       // A: K groups {kb..kb+7, kb+16..kb+23}
        const int kbB = (lane >> 4) * 8;       // B: K groups {kb..kb+7, kb+16..kb+23}
        #pragma unroll
        for (int i = 0; i < 2; ++i) {
            int m = wm * 32 + i * 16 + (lane & 15);
            union { v16bf v; v8bf h[2]; } af;
            af.h[0] = *(const v8bf*)&ldsA[m][kbA];
            af.h[1] = *(const v8bf*)&ldsA[m][kbA + 16];
            #pragma unroll
            for (int j = 0; j < 2; ++j) {
                int n = wn * 32 + j * 16 + (lane & 15);
                union { v16bf v; v8bf h[2]; } bfv;
                bfv.h[0] = *(const v8bf*)&ldsB[n][kbB];
                bfv.h[1] = *(const v8bf*)&ldsB[n][kbB + 16];
                acc[i][j] = __builtin_amdgcn_wmma_f32_16x16x32_bf16(
                    false, af.v, false, bfv.v, (short)0, acc[i][j], false, false);
            }
        }
        __syncthreads();
    }

    // Routed epilogue: q -> f32 [B,H,N,48]; k -> f32 [B,H,48,N]; v -> bf16 [B,H,N,48]
    #pragma unroll
    for (int i = 0; i < 2; ++i)
    #pragma unroll
    for (int j = 0; j < 2; ++j) {
        #pragma unroll
        for (int r = 0; r < 8; ++r) {
            int gm = tileM + wm * 32 + i * 16 + (lane >> 4) * 8 + r;
            int gn = tileN + wn * 32 + j * 16 + (lane & 15);
            float val = acc[i][j][r];
            int b     = gm >> 14;
            int n     = gm & (NTOK - 1);
            int which = gn / DIMD;
            int rem   = gn - which * DIMD;
            int h     = rem / CH;
            int c     = rem - h * CH;
            size_t bh = (size_t)(b * HEADS + h);
            if (which == 0)      q_buf[(bh * NTOK + n) * CH + c] = val;
            else if (which == 1) kT_buf[(bh * CH + c) * NTOK + n] = val;
            else                 v_buf[(bh * NTOK + n) * CH + c] = (__bf16)val;
        }
    }
}

// ===========================================================================
// K2: q softmax over C (48-wide rows) -> bf16 [B,H,N,64], cols 48..63 zeroed
// ===========================================================================
__global__ __launch_bounds__(256) void k_q_softmax(
    const float* __restrict__ q_buf, __bf16* __restrict__ q_pad)
{
    int lane = threadIdx.x & 31;
    int wave = threadIdx.x >> 5;
    size_t row = (size_t)blockIdx.x * 8 + wave;
    const float* src = q_buf + row * CH;

    float a0 = src[lane];
    float a1 = (lane < 16) ? src[32 + lane] : -INFINITY;
    float m = fmaxf(a0, a1);
    #pragma unroll
    for (int off = 16; off > 0; off >>= 1) m = fmaxf(m, __shfl_xor(m, off, 32));
    float e0 = __expf(a0 - m);
    float e1 = (lane < 16) ? __expf(a1 - m) : 0.0f;
    float s = e0 + e1;
    #pragma unroll
    for (int off = 16; off > 0; off >>= 1) s += __shfl_xor(s, off, 32);
    float inv = 1.0f / s;

    __bf16* dst = q_pad + row * 64;
    dst[lane]      = (__bf16)(e0 * inv);
    dst[32 + lane] = (lane < 16) ? (__bf16)(e1 * inv) : (__bf16)0.0f;
}

// ===========================================================================
// K3: k softmax stats over N: per (b,h,c) row of kT, compute (max, sum_exp)
// ===========================================================================
__global__ __launch_bounds__(256) void k_k_stats(
    const float* __restrict__ kT, float2* __restrict__ stats)
{
    __shared__ float red[256];
    size_t row = blockIdx.x;
    const float* src = kT + row * (size_t)NTOK;
    int t = threadIdx.x;

    float m = -INFINITY;
    for (int i = t; i < NTOK; i += 256) m = fmaxf(m, src[i]);
    red[t] = m; __syncthreads();
    for (int s = 128; s > 0; s >>= 1) { if (t < s) red[t] = fmaxf(red[t], red[t + s]); __syncthreads(); }
    m = red[0]; __syncthreads();

    float sum = 0.f;
    for (int i = t; i < NTOK; i += 256) sum += __expf(src[i] - m);
    red[t] = sum; __syncthreads();
    for (int s = 128; s > 0; s >>= 1) { if (t < s) red[t] += red[t + s]; __syncthreads(); }
    if (t == 0) stats[row] = make_float2(m, red[0]);
}

// ===========================================================================
// K4: context partials: ctx_part[bh,chunk] = sum_{n in chunk} exp(kT - max) * v
// grid (B*H, NCHUNK), block 128 = 4 waves; each wave owns 512 tokens of K.
// v slices staged per-wave with global_load_async_to_lds_b128 (ASYNCcnt).
// ===========================================================================
__global__ __launch_bounds__(128) void k_context(
    const float* __restrict__ kT, const __bf16* __restrict__ v_buf,
    const float2* __restrict__ stats, float* __restrict__ ctx_part)
{
    __shared__ float  cwork[4][CH * CH];         // per-wave partial (36 KB)
    __shared__ __bf16 vstage[4][32][CH];         // per-wave v slice (12 KB)
    int t = threadIdx.x, lane = t & 31, wave = t >> 5;
    int bh = blockIdx.x, chunk = blockIdx.y;

    float mrow[3];
    #pragma unroll
    for (int tm = 0; tm < 3; ++tm) mrow[tm] = stats[bh * CH + tm * 16 + (lane & 15)].x;

    const v8f vzero = {};
    v8f acc[3][3];
    for (int i = 0; i < 3; ++i) for (int j = 0; j < 3; ++j) acc[i][j] = vzero;

    const int kBase = chunk * CHUNK + wave * (CHUNK / 4);
    const float*  kTb = kT    + (size_t)bh * CH * NTOK;
    const __bf16* vb  = v_buf + (size_t)bh * NTOK * CH;
    const int kb2 = (lane >> 4) * 16;
    const uint32_t vstage_base = (uint32_t)(uintptr_t)(&vstage[wave][0][0]);

    for (int kk = 0; kk < CHUNK / 4; kk += 32) {
        int kOff = kBase + kk;
        // stage 32x48 bf16 v slice (3 KB, contiguous) into this wave's LDS buf
        const char* vsrc = (const char*)(vb + (size_t)kOff * CH);
        #pragma unroll
        for (int c = 0; c < 6; ++c) {
            int chnk = lane + 32 * c;            // 0..191 16B chunks
            async_b128_to_lds(vstage_base + (uint32_t)chnk * 16u, vsrc + chnk * 16);
        }
        // A = exp(kT[c, n] - max_c) fragments (overlaps with async copy)
        union { v16bf v; __bf16 e[16]; } af[3];
        #pragma unroll
        for (int tm = 0; tm < 3; ++tm) {
            int m = tm * 16 + (lane & 15);
            const float* rowp = kTb + (size_t)m * NTOK + kOff + (lane >> 4) * 16;
            float mx = mrow[tm];
            #pragma unroll
            for (int e = 0; e < 8; ++e) {
                af[tm].e[e]     = (__bf16)__expf(rowp[e]      - mx);
                af[tm].e[8 + e] = (__bf16)__expf(rowp[8 + e] - mx);
            }
        }
        wait_asynccnt0();                        // wave-private slice ready
        // B = v fragments from LDS, then 9 WMMAs
        #pragma unroll
        for (int tn = 0; tn < 3; ++tn) {
            union { v16bf v; __bf16 e[16]; } bfv;
            int col = tn * 16 + (lane & 15);
            #pragma unroll
            for (int e = 0; e < 16; ++e) bfv.e[e] = vstage[wave][kb2 + e][col];
            #pragma unroll
            for (int tm = 0; tm < 3; ++tm)
                acc[tm][tn] = __builtin_amdgcn_wmma_f32_16x16x32_bf16(
                    false, af[tm].v, false, bfv.v, (short)0, acc[tm][tn], false, false);
        }
    }

    #pragma unroll
    for (int tm = 0; tm < 3; ++tm)
    #pragma unroll
    for (int tn = 0; tn < 3; ++tn) {
        #pragma unroll
        for (int r = 0; r < 8; ++r) {
            int m = tm * 16 + (lane >> 4) * 8 + r;
            int n = tn * 16 + (lane & 15);
            cwork[wave][m * CH + n] = acc[tm][tn][r];
        }
    }
    __syncthreads();
    float* dst = ctx_part + ((size_t)bh * NCHUNK + chunk) * (CH * CH);
    for (int i = t; i < CH * CH; i += 128)
        dst[i] = (cwork[0][i] + cwork[1][i]) + (cwork[2][i] + cwork[3][i]);
}

// ===========================================================================
// K5: reduce ctx partials over chunks, scale row c by 1/sum_c, emit bf16
// ===========================================================================
__global__ __launch_bounds__(256) void k_ctx_reduce(
    const float* __restrict__ ctx_part, const float2* __restrict__ stats,
    __bf16* __restrict__ ctx_b)
{
    int idx = blockIdx.x * 256 + threadIdx.x;
    int bh  = idx / (64 * CH);
    int rem = idx - bh * (64 * CH);
    int c   = rem / CH;
    int d   = rem - c * CH;
    float val = 0.f;
    if (c < CH) {
        const float* p = ctx_part + (size_t)bh * NCHUNK * (CH * CH) + c * CH + d;
        float s = 0.f;
        #pragma unroll
        for (int ch = 0; ch < NCHUNK; ++ch) s += p[ch * (CH * CH)];
        val = s / stats[bh * CH + c].y;
    }
    ctx_b[idx] = (__bf16)val;
}

// ===========================================================================
// K6: attn = q_pad @ ctx   per (b,h): M=16384, N=48, K=64 (padded)
// ctx staged with async-to-LDS copies.
// ===========================================================================
__global__ __launch_bounds__(256) void k_attn_out(
    const __bf16* __restrict__ q_pad, const __bf16* __restrict__ ctx_b,
    __bf16* __restrict__ attn)
{
    __shared__ __bf16 ldsC[64 * CH];
    int t = threadIdx.x, lane = t & 31, wave = t >> 5;
    int bh = blockIdx.x;
    int mBase = blockIdx.y * 128;

    {
        const char* gsrc = (const char*)(ctx_b + (size_t)bh * 64 * CH);  // 384 x 16B
        const uint32_t lbase = (uint32_t)(uintptr_t)ldsC;
        for (int i = t; i < 384; i += 256)
            async_b128_to_lds(lbase + (uint32_t)i * 16u, gsrc + i * 16);
        wait_asynccnt0();
    }
    __syncthreads();
    asm volatile("" ::: "memory");

    const v8f vzero = {};
    v8f acc[3];
    for (int j = 0; j < 3; ++j) acc[j] = vzero;

    const __bf16* qb = q_pad + ((size_t)bh * NTOK + mBase + wave * 16) * 64;
    const int kbA = (lane >> 4) * 16;
    const int kb2 = (lane >> 4) * 16;

    #pragma unroll
    for (int kk = 0; kk < 64; kk += 32) {
        union { v16bf v; v8bf h[2]; } af;
        const __bf16* rowp = qb + (size_t)(lane & 15) * 64 + kk;
        af.h[0] = *(const v8bf*)(rowp + kbA);
        af.h[1] = *(const v8bf*)(rowp + kbA + 8);
        #pragma unroll
        for (int tn = 0; tn < 3; ++tn) {
            union { v16bf v; __bf16 e[16]; } bfv;
            int col = tn * 16 + (lane & 15);
            #pragma unroll
            for (int e = 0; e < 16; ++e) bfv.e[e] = ldsC[(kk + kb2 + e) * CH + col];
            acc[tn] = __builtin_amdgcn_wmma_f32_16x16x32_bf16(
                false, af.v, false, bfv.v, (short)0, acc[tn], false, false);
        }
    }

    int b = bh >> 3, h = bh & 7;
    #pragma unroll
    for (int tn = 0; tn < 3; ++tn) {
        #pragma unroll
        for (int r = 0; r < 8; ++r) {
            int m    = mBase + wave * 16 + (lane >> 4) * 8 + r;
            int dcol = tn * 16 + (lane & 15);
            attn[((size_t)b * NTOK + m) * DIMD + h * CH + dcol] = (__bf16)acc[tn][r];
        }
    }
}

// ===========================================================================
// K7: out = attn @ W_proj + b_proj   (M=131072, N=384, K=384) -> f32 d_out
// A tile (bf16, straight copy) moved by the Tensor Data Mover when available.
// ===========================================================================
__global__ __launch_bounds__(256) void k_proj(
    const __bf16* __restrict__ attn, const float* __restrict__ Wp,
    const float* __restrict__ bias, float* __restrict__ out)
{
    __shared__ __bf16 ldsA[128][40];   // 32-elem rows + 8-elem pad (TDM pad 16DW+4DW)
    __shared__ __bf16 ldsB[64][40];

    const int t    = threadIdx.x;
    const int lane = t & 31;
    const int wave = t >> 5;
    const int wm   = wave & 3;
    const int wn   = wave >> 2;
    const int tileM = blockIdx.x * 128;
    const int tileN = blockIdx.y * 64;

    const v8f vzero = {};
    v8f acc[2][2];
    for (int i = 0; i < 2; ++i) for (int j = 0; j < 2; ++j) acc[i][j] = vzero;

#if HAVE_TDM
    const uint32_t ldsA_base = (uint32_t)(uintptr_t)(&ldsA[0][0]);
#endif

    for (int k0 = 0; k0 < DIMD; k0 += 32) {
#if HAVE_TDM
        // A tile: TDM 2D descriptor, 32x128 bf16 tile, row stride 384,
        // LDS padding: 16 DWORDs data + 4 DWORDs pad => 40-elem LDS rows.
        if (wave == 0)
            tdm_load_2d_bf16(ldsA_base,
                             attn + (size_t)tileM * DIMD + k0,
                             /*tile_w=*/32, /*tile_h=*/128, /*row_stride=*/DIMD,
                             /*pad_interval(16DW)=*/3, /*pad_amount(4DW)=*/3, true);
#else
        #pragma unroll
        for (int q = 0; q < 2; ++q) {
            int fid  = t + 256 * q;
            int row  = fid >> 2;
            int colv = fid & 3;
            v8bf d = *(const v8bf*)(attn + (size_t)(tileM + row) * DIMD + k0 + colv * 8);
            *(v8bf*)&ldsA[row][colv * 8] = d;
        }
#endif
        // B tile: 32x64 f32 -> bf16, stored [n][k] (all 8 waves)
        float4 fb[2];
        #pragma unroll
        for (int q = 0; q < 2; ++q) {
            int fid  = t + 256 * q;
            int row  = fid >> 4;
            int colv = fid & 15;
            fb[q] = *(const float4*)(Wp + (size_t)(k0 + row) * DIMD + tileN + colv * 4);
        }
        #pragma unroll
        for (int q = 0; q < 2; ++q) {
            int fid  = t + 256 * q;
            int row  = fid >> 4;
            int colv = fid & 15;
            ldsB[colv * 4 + 0][row] = (__bf16)fb[q].x;
            ldsB[colv * 4 + 1][row] = (__bf16)fb[q].y;
            ldsB[colv * 4 + 2][row] = (__bf16)fb[q].z;
            ldsB[colv * 4 + 3][row] = (__bf16)fb[q].w;
        }
#if HAVE_TDM
        if (wave == 0) __builtin_amdgcn_s_wait_tensorcnt((short)0);
#endif
        __syncthreads();
        asm volatile("" ::: "memory");

        const int kbA = (lane >> 4) * 8;
        const int kbB = (lane >> 4) * 8;
        #pragma unroll
        for (int i = 0; i < 2; ++i) {
            int m = wm * 32 + i * 16 + (lane & 15);
            union { v16bf v; v8bf h[2]; } af;
            af.h[0] = *(const v8bf*)&ldsA[m][kbA];
            af.h[1] = *(const v8bf*)&ldsA[m][kbA + 16];
            #pragma unroll
            for (int j = 0; j < 2; ++j) {
                int n = wn * 32 + j * 16 + (lane & 15);
                union { v16bf v; v8bf h[2]; } bfv;
                bfv.h[0] = *(const v8bf*)&ldsB[n][kbB];
                bfv.h[1] = *(const v8bf*)&ldsB[n][kbB + 16];
                acc[i][j] = __builtin_amdgcn_wmma_f32_16x16x32_bf16(
                    false, af.v, false, bfv.v, (short)0, acc[i][j], false, false);
            }
        }
        __syncthreads();
    }

    #pragma unroll
    for (int i = 0; i < 2; ++i)
    #pragma unroll
    for (int j = 0; j < 2; ++j) {
        #pragma unroll
        for (int r = 0; r < 8; ++r) {
            size_t gm = (size_t)tileM + wm * 32 + i * 16 + (lane >> 4) * 8 + r;
            int    gn = tileN + wn * 32 + j * 16 + (lane & 15);
            out[gm * DIMD + gn] = acc[i][j][r] + bias[gn];
        }
    }
}

// ===========================================================================
extern "C" void kernel_launch(void* const* d_in, const int* in_sizes, int n_in,
                              void* d_out, int out_size, void* d_ws, size_t ws_size,
                              hipStream_t stream)
{
    (void)in_sizes; (void)n_in; (void)out_size; (void)ws_size;
    const float* x     = (const float*)d_in[0];
    const float* Wqkv  = (const float*)d_in[1];
    const float* Wproj = (const float*)d_in[2];
    const float* bproj = (const float*)d_in[3];
    float* out = (float*)d_out;

    char* ws = (char*)d_ws;
    float*  q_buf    = (float*)  (ws + OFF_Q);
    float*  kT_buf   = (float*)  (ws + OFF_KT);
    __bf16* v_buf    = (__bf16*) (ws + OFF_V);
    __bf16* q_pad    = (__bf16*) (ws + OFF_QPAD);
    float2* stats    = (float2*) (ws + OFF_STAT);
    float*  ctx_part = (float*)  (ws + OFF_CTXP);
    __bf16* ctx_b    = (__bf16*) (ws + OFF_CTXB);
    __bf16* attn     = (__bf16*) (ws + OFF_ATTN);

    k_qkv_gemm  <<<dim3(MTOT / 128, QKV3 / 64), 256, 0, stream>>>(x, Wqkv, q_buf, kT_buf, v_buf);
    k_q_softmax <<<dim3(BATCH * HEADS * NTOK / 8), 256, 0, stream>>>(q_buf, q_pad);
    k_k_stats   <<<dim3(BATCH * HEADS * CH), 256, 0, stream>>>(kT_buf, stats);
    k_context   <<<dim3(BATCH * HEADS, NCHUNK), 128, 0, stream>>>(kT_buf, v_buf, stats, ctx_part);
    k_ctx_reduce<<<dim3(BATCH * HEADS * 64 * CH / 256), 256, 0, stream>>>(ctx_part, stats, ctx_b);
    k_attn_out  <<<dim3(BATCH * HEADS, NTOK / 128), 256, 0, stream>>>(q_pad, ctx_b, attn);
    k_proj      <<<dim3(MTOT / 128, DIMD / 64), 256, 0, stream>>>(attn, Wproj, bproj, out);
}
